// AttentionRelativeBias_10952166605348
// MI455X (gfx1250) — matrix operand order repaired and checked by reading, not compiled, not run on hardware
//
#include <hip/hip_runtime.h>
#include <hip/hip_bf16.h>

typedef __attribute__((ext_vector_type(16))) __bf16       v16bf;
typedef __attribute__((ext_vector_type(8)))  float        v8f;
typedef __attribute__((ext_vector_type(8)))  unsigned int v8u;

#define NBATCH 128
#define SEQ    256
#define DIM    768
#define NHEAD  12
#define HDIM   64

#define WMMA_BF16(A,B,C) \
  __builtin_amdgcn_wmma_f32_16x16x32_bf16(false,(A),false,(B),(short)0,(C),false,false)

__device__ __forceinline__ unsigned short f2bf(float f) {
  return __builtin_bit_cast(unsigned short, (__bf16)f);
}

__device__ __forceinline__ v8f vzero8() {
  v8f z = {0.f,0.f,0.f,0.f,0.f,0.f,0.f,0.f};
  return z;
}

// LDS byte offset of a generic pointer to __shared__ memory (AS3->flat keeps
// the LDS offset in the low 32 bits).
__device__ __forceinline__ unsigned int lds_off(const void* p) {
  return (unsigned int)(unsigned long long)p;
}

// CDNA5 async global->LDS copy, 16 bytes per lane. Tracked by ASYNCcnt.
// GLOBAL_LOAD_ASYNC_TO_LDS_B128 VDST(lds addr), VADDR(64b global), SADDR=off
__device__ __forceinline__ void async_g2l_b128(unsigned int lds_byte, const void* gaddr) {
  unsigned long long ga = (unsigned long long)gaddr;
  asm volatile("global_load_async_to_lds_b128 %0, %1, off"
               :: "v"(lds_byte), "v"(ga) : "memory");
}

__device__ __forceinline__ void wait_async0() {
  asm volatile("s_wait_asynccnt 0x0" ::: "memory");
}

// A-matrix fragment: 16x32 bf16 (MxK), row-major in LDS, row stride `ld` (elements).
// ISA 7.12.2: lanes 0-15 M=lane, lanes 16-31 same M, K offset +8; VGPR j<4: K=2j(+h),
// j>=4: K=16+2(j-4)(+h).
__device__ __forceinline__ v16bf load_a_frag(const unsigned short* p, int ld, int lane) {
  int m = lane & 15, g = lane >> 4;
  v8u u;
#pragma unroll
  for (int j = 0; j < 8; ++j) {
    int kb = ((j & 4) << 2) + (g << 3) + ((j & 3) << 1);
    u[j] = *(const unsigned int*)(p + m * ld + kb);
  }
  return __builtin_bit_cast(v16bf, u);
}

// B-matrix fragment from N-major (transposed) storage [N][K], row stride `ld`.
// Same K order as A: lanes 16-31 K offset +8; VGPR j<4: K=2j(+h), j>=4: K=16+2(j-4)(+h).
__device__ __forceinline__ v16bf load_bt_frag(const unsigned short* p, int ld, int lane) {
  int n = lane & 15, g = lane >> 4;
  v8u u;
#pragma unroll
  for (int j = 0; j < 8; ++j) {
    int kb = ((j & 4) << 2) + (g << 3) + ((j & 3) << 1);
    u[j] = *(const unsigned int*)(p + n * ld + kb);
  }
  return __builtin_bit_cast(v16bf, u);
}

// ---------------------------------------------------------------------------
// Projection GEMM: Y = X @ W^T + b, fp32 in, bf16 out.
// layout 0: out[((b*H+h)*S+s)*DH+dh]   (Q, K)
// layout 1: out[((b*H+h)*DH+dh)*S+s]   (V transposed for P@V)
// Block: 128 threads (4 waves), 64x64 output tile, K-step 32.
// ---------------------------------------------------------------------------
__global__ void __launch_bounds__(128)
proj_kernel(const float* __restrict__ X, const float* __restrict__ Wt,
            const float* __restrict__ bias, unsigned short* __restrict__ out,
            int layout) {
  const int LDA = 40, LDB = 40;
  __shared__ unsigned short Al[64 * 40];
  __shared__ unsigned short Bl[64 * 40];

  int tid = threadIdx.x, lane = tid & 31, wave = tid >> 5;
  int bx = blockIdx.x;
  int m0 = (bx & 511) * 64;   // 512 M-tiles over 32768 rows
  int n0 = (bx >> 9) * 64;    // 12 N-tiles over 768 cols

  v8f acc[4];
#pragma unroll
  for (int t = 0; t < 4; ++t) acc[t] = vzero8();

  for (int k0 = 0; k0 < DIM; k0 += 32) {
    for (int l = tid; l < 2048; l += 128) {
      int r = l >> 5, c = l & 31;
      Al[r * LDA + c] = f2bf(X [(size_t)(m0 + r) * DIM + k0 + c]);
      Bl[r * LDB + c] = f2bf(Wt[(size_t)(n0 + r) * DIM + k0 + c]);
    }
    // Warm L2 for the next K slab (global_prefetch_b8): 64 rows x 2 lines.
    if (k0 + 32 < DIM) {
      int r = tid & 63, c = (tid >> 6) * 16;
      __builtin_prefetch(&X [(size_t)(m0 + r) * DIM + k0 + 32 + c], 0, 0);
      __builtin_prefetch(&Wt[(size_t)(n0 + r) * DIM + k0 + 32 + c], 0, 0);
    }
    __syncthreads();
    v16bf a = load_a_frag(Al + wave * 16 * LDA, LDA, lane);
#pragma unroll
    for (int t = 0; t < 4; ++t) {
      v16bf b = load_bt_frag(Bl + t * 16 * LDB, LDB, lane);
      acc[t] = WMMA_BF16(a, b, acc[t]);
    }
    __syncthreads();
  }

  int g = lane >> 4, nl = lane & 15;
#pragma unroll
  for (int t = 0; t < 4; ++t)
#pragma unroll
    for (int r = 0; r < 8; ++r) {
      int i = m0 + wave * 16 + g * 8 + r;  // global row in [0,32768)
      int o = n0 + t * 16 + nl;            // global col in [0,768)
      float vv = acc[t][r] + bias[o];
      int bb = i >> 8, s = i & 255;
      int hh = o >> 6, dh = o & 63;
      size_t off = (layout == 0)
        ? ((((size_t)bb * NHEAD + hh) * SEQ + s) * HDIM + dh)
        : ((((size_t)bb * NHEAD + hh) * HDIM + dh) * SEQ + s);
      out[off] = f2bf(vv);
    }
}

// ---------------------------------------------------------------------------
// Attention: one block per (b, h, 64-query strip). 256 threads = 8 waves.
// Staging via GLOBAL_LOAD_ASYNC_TO_LDS_B128 (ASYNCcnt path).
// Phase A: S = scale*Q@K^T + swin_bias  -> fp32 scores strip [64,256] in LDS
// Phase B: softmax over full row (256 keys)
// Phase C: ctx = P @ V  (P bf16, V^T staged N-major) -> bf16 [B,S,H,DH]
// ---------------------------------------------------------------------------
__global__ void __launch_bounds__(256)
attn_kernel(const unsigned short* __restrict__ qh, const unsigned short* __restrict__ kh,
            const unsigned short* __restrict__ vt, const float* __restrict__ bias_table,
            unsigned short* __restrict__ ctx) {
  extern __shared__ char smem[];
  const int LDQ = 72, LDK = 72, LDV = 264, LDS = 260, LDP = 264;
  unsigned short* Qs = (unsigned short*)smem;        //  64 x 72 bf16 (row = 144B, 16B-mult)
  unsigned short* Ks = Qs + 64 * LDQ;                // 256 x 72 bf16
  unsigned short* Vs = Ks + 256 * LDK;               //  64 x 264 bf16 (row = 528B, 16B-mult)
  float*          Sc = (float*)(Vs + 64 * LDV);      //  64 x 260 fp32
  unsigned short* Pl = (unsigned short*)(Sc + 64 * LDS); // 64 x 264 bf16
  float*        pmax = (float*)(Pl + 64 * LDP);      //  64 x 4
  float*        psum = pmax + 256;                   //  64 x 4

  int tid = threadIdx.x, lane = tid & 31, wave = tid >> 5;
  int bx = blockIdx.x;
  int bh = bx >> 2, qb = bx & 3;
  int b = bh / NHEAD, h = bh % NHEAD;
  int sq0 = qb * 64;

  const char* qsrc = (const char*)(qh + (((size_t)b * NHEAD + h) * SEQ + sq0) * HDIM);
  const char* ksrc = (const char*)(kh + ((size_t)b * NHEAD + h) * SEQ * HDIM);
  const char* vsrc = (const char*)(vt + ((size_t)b * NHEAD + h) * HDIM * SEQ);

  unsigned int qB = lds_off(Qs), kB = lds_off(Ks), vB = lds_off(Vs);

  // Q strip: 64x64 bf16 = 512 x 16B chunks
  for (int ch = tid; ch < 512; ch += 256) {
    int e = ch << 3, r = e >> 6, c = e & 63;
    async_g2l_b128(qB + (unsigned)(r * LDQ + c) * 2, qsrc + (size_t)ch * 16);
  }
  // K: 256x64 bf16 = 2048 chunks
  for (int ch = tid; ch < 2048; ch += 256) {
    int e = ch << 3, r = e >> 6, c = e & 63;
    async_g2l_b128(kB + (unsigned)(r * LDK + c) * 2, ksrc + (size_t)ch * 16);
  }
  // V^T: 64x256 bf16 = 2048 chunks
  for (int ch = tid; ch < 2048; ch += 256) {
    int e = ch << 3, r = e >> 8, c = e & 255;
    async_g2l_b128(vB + (unsigned)(r * LDV + c) * 2, vsrc + (size_t)ch * 16);
  }
  wait_async0();
  __syncthreads();

  // ---- Phase A: scores ----
  int mt = wave & 3, nh = wave >> 2;                 // wave -> (M-tile, N-half)
  int g = lane >> 4, nl = lane & 15;
  v8f sacc[8];
#pragma unroll
  for (int t = 0; t < 8; ++t) sacc[t] = vzero8();

  for (int kk = 0; kk < HDIM; kk += 32) {
    v16bf a = load_a_frag(Qs + mt * 16 * LDQ + kk, LDQ, lane);
#pragma unroll
    for (int t = 0; t < 8; ++t) {
      v16bf bfr = load_bt_frag(Ks + (nh * 8 + t) * 16 * LDK + kk, LDK, lane);
      sacc[t] = WMMA_BF16(a, bfr, sacc[t]);
    }
  }
#pragma unroll
  for (int t = 0; t < 8; ++t)
#pragma unroll
    for (int r = 0; r < 8; ++r) {
      int sq = mt * 16 + g * 8 + r;                  // query idx within strip (== window idx)
      int sk = (nh * 8 + t) * 16 + nl;               // key idx in [0,256)
      int km = sk & 63;                              // bias tiles repeat every 64
      int qi = sq >> 3, qj = sq & 7, ki = km >> 3, kj = km & 7;
      int idx = (qi - ki + 7) * 15 + (qj - kj + 7);  // (2W-1)=15 relative index
      Sc[sq * LDS + sk] = sacc[t][r] * 0.125f + bias_table[idx * NHEAD + h];
    }
  __syncthreads();

  // ---- Phase B: softmax (4 threads per row) ----
  int rr = tid >> 2, q4 = tid & 3, c0 = q4 * 64;
  float m = -3.4e38f;
  for (int c = c0; c < c0 + 64; ++c) m = fmaxf(m, Sc[rr * LDS + c]);
  pmax[rr * 4 + q4] = m;
  __syncthreads();
  float rm = fmaxf(fmaxf(pmax[rr * 4], pmax[rr * 4 + 1]),
                   fmaxf(pmax[rr * 4 + 2], pmax[rr * 4 + 3]));
  float s = 0.f;
  for (int c = c0; c < c0 + 64; ++c) {
    float e = __expf(Sc[rr * LDS + c] - rm);
    Sc[rr * LDS + c] = e;
    s += e;
  }
  psum[rr * 4 + q4] = s;
  __syncthreads();
  float inv = 1.f / (psum[rr * 4] + psum[rr * 4 + 1] + psum[rr * 4 + 2] + psum[rr * 4 + 3]);
  for (int c = c0; c < c0 + 64; ++c)
    Pl[rr * LDP + c] = f2bf(Sc[rr * LDS + c] * inv);
  __syncthreads();

  // ---- Phase C: ctx = P @ V ----
  v8f oacc[2];
  oacc[0] = vzero8(); oacc[1] = vzero8();
  int nt0 = nh * 2;
  for (int j = 0; j < SEQ; j += 32) {
    v16bf a = load_a_frag(Pl + mt * 16 * LDP + j, LDP, lane);
#pragma unroll
    for (int t = 0; t < 2; ++t) {
      v16bf bfr = load_bt_frag(Vs + (nt0 + t) * 16 * LDV + j, LDV, lane);
      oacc[t] = WMMA_BF16(a, bfr, oacc[t]);
    }
  }
#pragma unroll
  for (int t = 0; t < 2; ++t)
#pragma unroll
    for (int r = 0; r < 8; ++r) {
      int sq = mt * 16 + g * 8 + r;
      int d  = (nt0 + t) * 16 + nl;
      size_t off = (((size_t)b * SEQ + sq0 + sq) * NHEAD + h) * HDIM + d;
      ctx[off] = f2bf(oacc[t][r]);
    }
}

// ---------------------------------------------------------------------------
// Output projection: out = ctx(bf16) @ Wo^T + bo, fp32 out [B,S,D].
// A tile staged with async global->LDS (already bf16); W converted via VALU.
// ---------------------------------------------------------------------------
__global__ void __launch_bounds__(128)
outproj_kernel(const unsigned short* __restrict__ A, const float* __restrict__ Wt,
               const float* __restrict__ bias, float* __restrict__ out) {
  const int LDA = 40, LDB = 40;                      // row = 80B, 16B-multiple
  __shared__ unsigned short Al[64 * 40];
  __shared__ unsigned short Bl[64 * 40];

  int tid = threadIdx.x, lane = tid & 31, wave = tid >> 5;
  int bx = blockIdx.x;
  int m0 = (bx & 511) * 64;
  int n0 = (bx >> 9) * 64;
  unsigned int aB = lds_off(Al);

  v8f acc[4];
#pragma unroll
  for (int t = 0; t < 4; ++t) acc[t] = vzero8();

  for (int k0 = 0; k0 < DIM; k0 += 32) {
    // A tile 64x32 bf16 = 256 x 16B chunks, async to LDS
    for (int ch = tid; ch < 256; ch += 128) {
      int e = ch << 3, r = e >> 5, c = e & 31;
      async_g2l_b128(aB + (unsigned)(r * LDA + c) * 2,
                     (const char*)(A + (size_t)(m0 + r) * DIM + k0 + c));
    }
    for (int l = tid; l < 2048; l += 128) {          // W fp32 -> bf16
      int r = l >> 5, c = l & 31;
      Bl[r * LDB + c] = f2bf(Wt[(size_t)(n0 + r) * DIM + k0 + c]);
    }
    if (k0 + 32 < DIM) {
      int r = tid & 63, c = (tid >> 6) * 16;
      __builtin_prefetch(&Wt[(size_t)(n0 + r) * DIM + k0 + 32 + c], 0, 0);
    }
    wait_async0();
    __syncthreads();
    v16bf a = load_a_frag(Al + wave * 16 * LDA, LDA, lane);
#pragma unroll
    for (int t = 0; t < 4; ++t) {
      v16bf b = load_bt_frag(Bl + t * 16 * LDB, LDB, lane);
      acc[t] = WMMA_BF16(a, b, acc[t]);
    }
    __syncthreads();
  }

  int g = lane >> 4, nl = lane & 15;
#pragma unroll
  for (int t = 0; t < 4; ++t)
#pragma unroll
    for (int r = 0; r < 8; ++r) {
      int i = m0 + wave * 16 + g * 8 + r;
      int o = n0 + t * 16 + nl;
      out[(size_t)i * DIM + o] = acc[t][r] + bias[o];
    }
}

extern "C" void kernel_launch(void* const* d_in, const int* in_sizes, int n_in,
                              void* d_out, int out_size, void* d_ws, size_t ws_size,
                              hipStream_t stream) {
  (void)in_sizes; (void)n_in; (void)out_size; (void)ws_size;
  const float* q  = (const float*)d_in[0];
  const float* k  = (const float*)d_in[1];
  const float* v  = (const float*)d_in[2];
  const float* Wq = (const float*)d_in[3];
  const float* bq = (const float*)d_in[4];
  const float* Wk = (const float*)d_in[5];
  const float* bk = (const float*)d_in[6];
  const float* Wv = (const float*)d_in[7];
  const float* bv = (const float*)d_in[8];
  const float* Wo = (const float*)d_in[9];
  const float* bo = (const float*)d_in[10];
  const float* bt = (const float*)d_in[11];
  float* out = (float*)d_out;

  unsigned short* ws = (unsigned short*)d_ws;
  const size_t NH = (size_t)NBATCH * NHEAD * SEQ * HDIM;   // 25,165,824 bf16 each
  unsigned short* qhp  = ws;
  unsigned short* khp  = ws + NH;
  unsigned short* vtp  = ws + 2 * NH;
  unsigned short* ctxp = ws + 3 * NH;

  dim3 gproj(6144), bproj(128);
  proj_kernel<<<gproj, bproj, 0, stream>>>(q, Wq, bq, qhp, 0);
  proj_kernel<<<gproj, bproj, 0, stream>>>(k, Wk, bk, khp, 0);
  proj_kernel<<<gproj, bproj, 0, stream>>>(v, Wv, bv, vtp, 1);

  size_t smem = (size_t)(64 * 72 + 256 * 72 + 64 * 264) * 2  // Q,K,V^T bf16
              + (size_t)64 * 260 * 4                          // scores fp32
              + (size_t)64 * 264 * 2                          // P bf16
              + 512 * 4;                                      // pmax/psum
  attn_kernel<<<dim3(NBATCH * NHEAD * 4), dim3(256), smem, stream>>>(qhp, khp, vtp, bt, ctxp);

  outproj_kernel<<<gproj, bproj, 0, stream>>>(ctxp, Wo, bo, out);
}
